// ResidualDiscriminator_77309411328716
// MI455X (gfx1250) — compile-verified, matrix operand order checked by reading, not run on hardware
//
#include <hip/hip_runtime.h>
#include <hip/hip_bf16.h>

typedef __attribute__((ext_vector_type(16))) _Float16 v16h;
typedef __attribute__((ext_vector_type(8)))  _Float16 v8h;
typedef __attribute__((ext_vector_type(8)))  float    v8f;

// ---------------- workspace layout (units: _Float16 elements) ----------------
// padded f16 weights, row-major [Npad][Kpad], zero padded
#define SZ_W0   (112 * 64)      // lin0: N=100->112, K=64
#define SZ_W1   (208 * 128)     // block w1: N=200->208, K=100->128
#define SZ_W2   (112 * 224)     // block w2: N=100->112, K=200->224
#define SZ_WMID (64 * 128)      // lin1: N=50->64, K=100->128
#define SZ_WOUT (16 * 64)       // lin2: N=1->16,  K=50->64
#define OFF_W0   0
#define OFF_BLK  (OFF_W0 + SZ_W0)
#define OFF_WMID (OFF_BLK + 3 * (SZ_W1 + SZ_W2))
#define OFF_WOUT (OFF_WMID + SZ_WMID)
#define WS_HALVES (OFF_WOUT + SZ_WOUT)

// ---------------- weight prep: f32 [N][K] -> padded f16 [Npad][Kpad] --------
struct PrepDesc { const float* src; _Float16* dst; int N, K, Npad, Kpad; };
struct PrepArgs { PrepDesc d[9]; };

__global__ void prep_weights_kernel(PrepArgs a) {
    PrepDesc d = a.d[blockIdx.x];
    int total = d.Npad * d.Kpad;
    for (int i = threadIdx.x; i < total; i += blockDim.x) {
        int n = i / d.Kpad;
        int k = i - n * d.Kpad;
        float v = (n < d.N && k < d.K) ? d.src[n * d.K + k] : 0.0f;
        d.dst[i] = (_Float16)v;
    }
}

// ---------------- fragment loader --------------------------------------------
// 16-bit A/B fragment for one 16x32 (or 32x16) tile, K-major per lane.
// lanes 0-15: elems = K[kbase+0..7] ++ K[kbase+16..23]
// lanes16-31: elems = K[kbase+8..15] ++ K[kbase+24..31]
__device__ __forceinline__ v16h load_frag(const _Float16* base, int stride,
                                          int kbase, int lane) {
    const int r   = lane & 15;
    const int off = (lane >> 4) << 3;
    const _Float16* p = base + r * stride + kbase + off;
    v8h lo = *(const v8h*)(p);
    v8h hi = *(const v8h*)(p + 16);
    v16h out;
#pragma unroll
    for (int i = 0; i < 8; ++i) { out[i] = lo[i]; out[i + 8] = hi[i]; }
    return out;
}

// ---------------- GEMM: acc[NT] (16x16 f32 tiles) = act @ W^T + bias ---------
template <int NT, int KT>
__device__ __forceinline__ void gemm_tiles(v8f (&acc)[NT],
                                           const _Float16* act, int astride,
                                           const _Float16* W, int wstride,
                                           const float* bias, int realN, int lane) {
    const int ncol = lane & 15;
#pragma unroll
    for (int t = 0; t < NT; ++t) {
        int n = t * 16 + ncol;
        float bb = (n < realN) ? bias[n] : 0.0f;
#pragma unroll
        for (int r = 0; r < 8; ++r) acc[t][r] = bb;
    }
#pragma unroll
    for (int kt = 0; kt < KT; ++kt) {
        v16h a = load_frag(act, astride, kt * 32, lane);
#pragma unroll
        for (int t = 0; t < NT; ++t) {
            v16h b = load_frag(W + t * 16 * wstride, wstride, kt * 32, lane);
            acc[t] = __builtin_amdgcn_wmma_f32_16x16x32_f16(
                false, a, false, b, (short)0, acc[t], false, false);
        }
    }
}

// ---------------- fused LayerNorm on C/D register layout ---------------------
// 16-lane-group XOR butterfly via DS_SWIZZLE_B32 group-of-32 mode:
// offset = {1'b0, xor_mask[4:0]=XOR, or_mask[4:0]=0, and_mask[4:0]=0x1f}
template <int XOR>
__device__ __forceinline__ float swz_xor_add(float v) {
    int o = __builtin_amdgcn_ds_swizzle(__float_as_int(v), (XOR << 10) | 0x1f);
    return v + __int_as_float(o);
}

__device__ __forceinline__ float group16_sum(float v) {
    v = swz_xor_add<1>(v);
    v = swz_xor_add<2>(v);
    v = swz_xor_add<4>(v);
    v = swz_xor_add<8>(v);
    return v;
}

template <int NT>
__device__ __forceinline__ void layernorm_tiles(v8f (&acc)[NT], int realN,
                                                const float* g, const float* b,
                                                int lane) {
    const int ncol = lane & 15;
    float s[8], q[8];
#pragma unroll
    for (int r = 0; r < 8; ++r) { s[r] = 0.0f; q[r] = 0.0f; }
#pragma unroll
    for (int t = 0; t < NT; ++t) {
        int n = t * 16 + ncol;
        float valid = (n < realN) ? 1.0f : 0.0f;
#pragma unroll
        for (int r = 0; r < 8; ++r) {
            float v = acc[t][r] * valid;
            s[r] += v;
            q[r] += v * v;
        }
    }
    float mu[8], rs[8];
    const float invn = 1.0f / (float)realN;
#pragma unroll
    for (int r = 0; r < 8; ++r) {
        float su = group16_sum(s[r]);
        float sq = group16_sum(q[r]);
        float m  = su * invn;
        float var = sq * invn - m * m;
        mu[r] = m;
        rs[r] = rsqrtf(var + 1e-5f);
    }
#pragma unroll
    for (int t = 0; t < NT; ++t) {
        int n = t * 16 + ncol;
        bool valid = (n < realN);
        float gg = valid ? g[n] : 0.0f;
        float bb = valid ? b[n] : 0.0f;
#pragma unroll
        for (int r = 0; r < 8; ++r)
            acc[t][r] = (acc[t][r] - mu[r]) * rs[r] * gg + bb;
    }
}

template <int NT>
__device__ __forceinline__ void prelu_tiles(v8f (&acc)[NT], float a) {
#pragma unroll
    for (int t = 0; t < NT; ++t)
#pragma unroll
        for (int r = 0; r < 8; ++r) {
            float v = acc[t][r];
            acc[t][r] = (v >= 0.0f) ? v : a * v;
        }
}

// C/D layout -> row-major f16 LDS staging (wave-private, no barrier needed)
template <int NT>
__device__ __forceinline__ void store_act(const v8f (&acc)[NT], _Float16* buf,
                                          int stride, int lane) {
    const int ncol = lane & 15;
    const int hi8  = (lane >> 4) << 3;
#pragma unroll
    for (int t = 0; t < NT; ++t) {
        int n = t * 16 + ncol;
#pragma unroll
        for (int r = 0; r < 8; ++r)
            buf[(r + hi8) * stride + n] = (_Float16)acc[t][r];
    }
}

// ---------------- fused network kernel ---------------------------------------
struct BlkParams {
    const _Float16 *w1, *w2;
    const float *b1, *g1, *bln1, *a, *b2, *g2, *bln2, *lns_g, *lns_b;
};
struct NetParams {
    const float* x;
    float* out;
    const _Float16* w0;
    const float *b0, *g0, *bb0, *a0;
    BlkParams blk[3];
    const _Float16* wmid;
    const float *bmid, *gmid, *bbmid, *a1;
    const _Float16* wout;
    const float* bout;
};

#define WAVES_PER_BLOCK 4
#define SA 224   // bufA stride (halves): fits K=224
#define SB 128   // bufB stride (halves): fits K=128

__global__ __launch_bounds__(WAVES_PER_BLOCK * 32)
void fused_mlp_kernel(NetParams p) {
    __shared__ _Float16 shA[WAVES_PER_BLOCK][16 * SA];
    __shared__ _Float16 shB[WAVES_PER_BLOCK][16 * SB];

    const int lane = threadIdx.x & 31;
    const int wave = threadIdx.x >> 5;
    _Float16* bufA = shA[wave];
    _Float16* bufB = shB[wave];

    // hint: weights are hot for the whole grid — pull them toward the caches
    if (threadIdx.x == 0) {
        __builtin_prefetch(p.w0, 0, 1);
        __builtin_prefetch(p.blk[0].w1, 0, 1);
    }

    // one-time zero fill of both staging buffers (covers all K padding)
    {
        v8h z = {};
        v8h* a8 = (v8h*)bufA;
        v8h* b8 = (v8h*)bufB;
#pragma unroll
        for (int i = 0; i < (16 * SA / 8) / 32; ++i) a8[lane + 32 * i] = z;
#pragma unroll
        for (int i = 0; i < (16 * SB / 8) / 32; ++i) b8[lane + 32 * i] = z;
    }

    const long row0 = ((long)blockIdx.x * WAVES_PER_BLOCK + wave) * 16;

    // ---- stage input tile: 16 rows x 64 f32 -> f16 LDS (bufA, stride SA) ----
    {
        const int m  = lane & 15;
        const int c0 = (lane >> 4) * 32;
        const float* xs = p.x + (row0 + m) * 64 + c0;
#pragma unroll
        for (int c = 0; c < 32; c += 8) {
            float4 f0 = *(const float4*)(xs + c);
            float4 f1 = *(const float4*)(xs + c + 4);
            v8h h;
            h[0] = (_Float16)f0.x; h[1] = (_Float16)f0.y;
            h[2] = (_Float16)f0.z; h[3] = (_Float16)f0.w;
            h[4] = (_Float16)f1.x; h[5] = (_Float16)f1.y;
            h[6] = (_Float16)f1.z; h[7] = (_Float16)f1.w;
            *(v8h*)(bufA + m * SA + c0 + c) = h;
        }
    }

    // ---- lin0 [64 -> 100(pad112)] + LN + PReLU ----
    v8f h0[7];
    gemm_tiles<7, 2>(h0, bufA, SA, p.w0, 64, p.b0, 100, lane);
    layernorm_tiles<7>(h0, 100, p.g0, p.bb0, lane);
    prelu_tiles<7>(h0, p.a0[0]);
    store_act<7>(h0, bufB, SB, lane);   // K staging for blocks (pad to 128)

    // ---- 3 residual blocks ----
    for (int i = 0; i < 3; ++i) {
        const BlkParams bp = p.blk[i];

        v8f t1[13];  // [100(pad128) -> 200(pad208)]
        gemm_tiles<13, 4>(t1, bufB, SB, bp.w1, 128, bp.b1, 200, lane);
        layernorm_tiles<13>(t1, 200, bp.g1, bp.bln1, lane);
        prelu_tiles<13>(t1, bp.a[0]);
        store_act<13>(t1, bufA, SA, lane);   // pad to K=224

        v8f t2[7];   // [200(pad224) -> 100(pad112)]
        gemm_tiles<7, 7>(t2, bufA, SA, bp.w2, 224, bp.b2, 100, lane);
        layernorm_tiles<7>(t2, 100, bp.g2, bp.bln2, lane);

        // residual add (kept in registers) + ReLU
#pragma unroll
        for (int t = 0; t < 7; ++t)
#pragma unroll
            for (int r = 0; r < 8; ++r) {
                float v = t2[t][r] + h0[t][r];
                t2[t][r] = (v > 0.0f) ? v : 0.0f;
            }

        // outer per-block LayerNorm
        layernorm_tiles<7>(t2, 100, bp.lns_g, bp.lns_b, lane);

#pragma unroll
        for (int t = 0; t < 7; ++t) h0[t] = t2[t];
        store_act<7>(t2, bufB, SB, lane);
    }

    // ---- lin1 [100(pad128) -> 50(pad64)] + LN + PReLU ----
    v8f m4[4];
    gemm_tiles<4, 4>(m4, bufB, SB, p.wmid, 128, p.bmid, 50, lane);
    layernorm_tiles<4>(m4, 50, p.gmid, p.bbmid, lane);
    prelu_tiles<4>(m4, p.a1[0]);
    store_act<4>(m4, bufA, SA, lane);    // cols 0..63, zeros in 50..63 already

    // ---- lin2 [50(pad64) -> 1(pad16)] + sigmoid ----
    v8f o1[1];
    gemm_tiles<1, 2>(o1, bufA, SA, p.wout, 64, p.bout, 1, lane);

    if ((lane & 15) == 0) {
        const int hi8 = (lane >> 4) << 3;
#pragma unroll
        for (int r = 0; r < 8; ++r) {
            float v = o1[0][r];
            p.out[row0 + hi8 + r] = 1.0f / (1.0f + __expf(-v));
        }
    }
}

// ---------------- host launcher ----------------------------------------------
extern "C" void kernel_launch(void* const* d_in, const int* in_sizes, int n_in,
                              void* d_out, int out_size, void* d_ws, size_t ws_size,
                              hipStream_t stream) {
    (void)in_sizes; (void)n_in; (void)out_size; (void)ws_size;

    // input index map (setup_inputs insertion order):
    // 0:x 1:lin0_w 2:lin0_b 3:ln0_g 4:ln0_b 5:prelu0
    // blocks i (base 6+9i): w1 b1 g1 bln1 a w2 b2 g2 bln2
    // 33..35: lns_g[0..2]  36..38: lns_b[0..2]
    // 39:lin1_w 40:lin1_b 41:ln1_g 42:ln1_b 43:prelu1 44:lin2_w 45:lin2_b
    _Float16* ws = (_Float16*)d_ws;

    _Float16* w0p   = ws + OFF_W0;
    _Float16* w1p[3], *w2p[3];
    for (int i = 0; i < 3; ++i) {
        w1p[i] = ws + OFF_BLK + i * (SZ_W1 + SZ_W2);
        w2p[i] = w1p[i] + SZ_W1;
    }
    _Float16* wmidp = ws + OFF_WMID;
    _Float16* woutp = ws + OFF_WOUT;

    PrepArgs pa;
    pa.d[0] = { (const float*)d_in[1],  w0p,   100,  64, 112,  64 };
    for (int i = 0; i < 3; ++i) {
        pa.d[1 + 2 * i] = { (const float*)d_in[6 + 9 * i],  w1p[i], 200, 100, 208, 128 };
        pa.d[2 + 2 * i] = { (const float*)d_in[11 + 9 * i], w2p[i], 100, 200, 112, 224 };
    }
    pa.d[7] = { (const float*)d_in[39], wmidp,  50, 100,  64, 128 };
    pa.d[8] = { (const float*)d_in[44], woutp,   1,  50,  16,  64 };
    prep_weights_kernel<<<9, 256, 0, stream>>>(pa);

    NetParams np;
    np.x   = (const float*)d_in[0];
    np.out = (float*)d_out;
    np.w0  = w0p;
    np.b0  = (const float*)d_in[2];
    np.g0  = (const float*)d_in[3];
    np.bb0 = (const float*)d_in[4];
    np.a0  = (const float*)d_in[5];
    for (int i = 0; i < 3; ++i) {
        int base = 6 + 9 * i;
        np.blk[i].w1    = w1p[i];
        np.blk[i].b1    = (const float*)d_in[base + 1];
        np.blk[i].g1    = (const float*)d_in[base + 2];
        np.blk[i].bln1  = (const float*)d_in[base + 3];
        np.blk[i].a     = (const float*)d_in[base + 4];
        np.blk[i].w2    = w2p[i];
        np.blk[i].b2    = (const float*)d_in[base + 6];
        np.blk[i].g2    = (const float*)d_in[base + 7];
        np.blk[i].bln2  = (const float*)d_in[base + 8];
        np.blk[i].lns_g = (const float*)d_in[33 + i];
        np.blk[i].lns_b = (const float*)d_in[36 + i];
    }
    np.wmid  = wmidp;
    np.bmid  = (const float*)d_in[40];
    np.gmid  = (const float*)d_in[41];
    np.bbmid = (const float*)d_in[42];
    np.a1    = (const float*)d_in[43];
    np.wout  = woutp;
    np.bout  = (const float*)d_in[45];

    const int B = in_sizes[0] / 64;                       // 262144 rows
    const int rows_per_block = WAVES_PER_BLOCK * 16;      // 64
    fused_mlp_kernel<<<B / rows_per_block, WAVES_PER_BLOCK * 32, 0, stream>>>(np);
}
